// MTCNN_7636451852923
// MI455X (gfx1250) — matrix operand order repaired and checked by reading, not compiled, not run on hardware
//
#include <hip/hip_runtime.h>
#include <hip/hip_bf16.h>

typedef __attribute__((ext_vector_type(16))) _Float16 v16h;
typedef __attribute__((ext_vector_type(8)))  _Float16 v8h;
typedef __attribute__((ext_vector_type(8)))  float    v8f;

#define DEVINL __device__ __forceinline__

// ---------------------------------------------------------------------------
// Weight prep: w (Cout x K, f32, row-major) -> wt (KPAD x Cout, f16), with
// zero padding for k >= K. WMMA B-fragments are gathered down k with stride Cout.
// ---------------------------------------------------------------------------
__global__ __launch_bounds__(256) void prep_weights(
    const float* __restrict__ w, _Float16* __restrict__ wt,
    int Cout, int K, int KPAD)
{
    int idx = blockIdx.x * blockDim.x + threadIdx.x;
    int total = KPAD * Cout;
    if (idx >= total) return;
    int k  = idx / Cout;
    int co = idx - k * Cout;
    wt[idx] = (k < K) ? (_Float16)w[(size_t)co * K + k] : (_Float16)0.0f;
}

// ---------------------------------------------------------------------------
// Fused: fixed_image_standardization + conv KHxKW (VALID) + bias + PReLU +
// ceil-mode maxpool (PK, stride PS). Weights cached in LDS. f32 accumulate,
// f16 output (all intermediates are stored f16 to halve HBM traffic).
// ---------------------------------------------------------------------------
template<int CIN, int KH, int KW, int COUT, int PK, int PS>
__global__ __launch_bounds__(256) void conv_std_prelu_pool(
    const float* __restrict__ in, const float* __restrict__ w,
    const float* __restrict__ bias, const float* __restrict__ alpha,
    _Float16* __restrict__ out,
    int N, int Hin, int Win, int Hc, int Wc, int Hp, int Wp)
{
    constexpr int KSZ = CIN * KH * KW;
    __shared__ float sw[COUT * KSZ];
    __shared__ float sb[COUT];
    __shared__ float sa[COUT];
    for (int i = threadIdx.x; i < COUT * KSZ; i += blockDim.x) sw[i] = w[i];
    if (threadIdx.x < COUT) { sb[threadIdx.x] = bias[threadIdx.x]; sa[threadIdx.x] = alpha[threadIdx.x]; }
    __syncthreads();

    int idx = blockIdx.x * blockDim.x + threadIdx.x;
    int total = N * Hp * Wp;
    if (idx >= total) return;
    int n   = idx / (Hp * Wp);
    int rem = idx - n * (Hp * Wp);
    int hp  = rem / Wp;
    int wp  = rem - hp * Wp;

    float best[COUT];
#pragma unroll
    for (int co = 0; co < COUT; ++co) best[co] = -3.4e38f;

    for (int ph = 0; ph < PK; ++ph) {
        int hc = hp * PS + ph;
        if (hc >= Hc) break;
        for (int pw = 0; pw < PK; ++pw) {
            int wc = wp * PS + pw;
            if (wc >= Wc) break;
            float acc[COUT];
#pragma unroll
            for (int co = 0; co < COUT; ++co) acc[co] = sb[co];
            const float* ip = in + ((size_t)n * CIN * Hin + hc) * Win + wc;
#pragma unroll
            for (int ci = 0; ci < CIN; ++ci)
#pragma unroll
            for (int kh = 0; kh < KH; ++kh)
#pragma unroll
            for (int kw = 0; kw < KW; ++kw) {
                float xv = (ip[((size_t)ci * Hin + kh) * Win + kw] - 127.5f) * 0.0078125f;
                const int widx = ci * KH * KW + kh * KW + kw;
#pragma unroll
                for (int co = 0; co < COUT; ++co)
                    acc[co] = fmaf(sw[co * KSZ + widx], xv, acc[co]);
            }
#pragma unroll
            for (int co = 0; co < COUT; ++co) best[co] = fmaxf(best[co], acc[co]);
        }
    }
#pragma unroll
    for (int co = 0; co < COUT; ++co) {
        float v = best[co];
        v = v > 0.f ? v : sa[co] * v;
        out[((size_t)(n * COUT + co) * Hp + hp) * Wp + wp] = (_Float16)v;
    }
}

// ---------------------------------------------------------------------------
// Implicit-GEMM conv via WMMA f16 (f32 accum) + bias + PReLU.
// f16 in / f16 out. Wave = one 16-row M tile x NTG 16-col N tiles.
// B fragments: strided gathers (stride COUT) from pre-transposed f16 weights.
// EXEC stays all-ones: tails handled by clamping (duplicate identical writes).
// ---------------------------------------------------------------------------
template<int CIN, int KH, int KW>
DEVINL _Float16 ldA_elem(const _Float16* inb, int Hin, int Win, int k)
{
    constexpr int K = CIN * KH * KW;
    if (k >= K) return (_Float16)0.0f;
    int ci = k / (KH * KW);
    int r  = k - ci * (KH * KW);
    int kh = r / KW;
    int kw = r - kh * KW;
    return inb[((size_t)ci * Hin + kh) * Win + kw];
}

template<int CIN, int KH, int KW, int COUT, int NTG>
__global__ __launch_bounds__(256) void conv_wmma_prelu(
    const _Float16* __restrict__ in, const _Float16* __restrict__ wt, // (KPAD, COUT)
    const float* __restrict__ bias, const float* __restrict__ alpha,
    _Float16* __restrict__ out, int N, int Hin, int Win, int Hc, int Wc)
{
    constexpr int K    = CIN * KH * KW;
    constexpr int KPAD = (K + 31) & ~31;
    constexpr int NG   = (COUT / 16) / NTG;

    int wave = threadIdx.x >> 5;
    int lane = threadIdx.x & 31;
    int hi   = lane >> 4;
    int l15  = lane & 15;

    int M  = N * Hc * Wc;
    int mt = (M + 15) >> 4;
    int totalTiles = mt * NG;
    int tile = blockIdx.x * 8 + wave;
    if (tile >= totalTiles) tile = totalTiles - 1;
    int tm = tile / NG;
    int tg = tile - tm * NG;

    int m = tm * 16 + l15;
    if (m >= M) m = M - 1;
    int hw  = Hc * Wc;
    int n   = m / hw;
    int rem = m - n * hw;
    int ho  = rem / Wc;
    int wo  = rem - ho * Wc;
    const _Float16* inb = in + ((size_t)(n * CIN) * Hin + ho) * Win + wo;

    v8f acc[NTG];
#pragma unroll
    for (int t = 0; t < NTG; ++t) acc[t] = (v8f){};

    for (int kc = 0; kc < KPAD; kc += 32) {
        v16h a;
        int ka = kc + hi * 8;   // ISA A layout: lanes 16-31 hold the +8 K block
#pragma unroll
        for (int i = 0; i < 8; ++i) {
            a[i]     = ldA_elem<CIN, KH, KW>(inb, Hin, Win, ka + i);
            a[i + 8] = ldA_elem<CIN, KH, KW>(inb, Hin, Win, ka + 16 + i);
        }
        const _Float16* wrow = wt + (size_t)(kc + hi * 8) * COUT + tg * (NTG * 16) + l15;
#pragma unroll
        for (int t = 0; t < NTG; ++t) {
            v16h b;   // column l15, K in table order (strided gather)
#pragma unroll
            for (int i = 0; i < 8; ++i) {
                b[i]     = wrow[(size_t)i * COUT + t * 16];
                b[i + 8] = wrow[(size_t)(16 + i) * COUT + t * 16];
            }
            acc[t] = __builtin_amdgcn_wmma_f32_16x16x32_f16(false, a, false, b,
                                                            (short)0, acc[t], false, false);
        }
    }

#pragma unroll
    for (int t = 0; t < NTG; ++t) {
        int co = (tg * NTG + t) * 16 + l15;
        float bn = bias[co], an = alpha[co];
#pragma unroll
        for (int r = 0; r < 8; ++r) {
            int mm = tm * 16 + (hi << 3) + r;
            if (mm >= M) mm = M - 1;
            int n2   = mm / hw;
            int rem2 = mm - n2 * hw;
            int ho2  = rem2 / Wc;
            int wo2  = rem2 - ho2 * Wc;
            float v = acc[t][r] + bn;
            v = v > 0.f ? v : an * v;
            out[((size_t)(n2 * COUT + co) * Hc + ho2) * Wc + wo2] = (_Float16)v;
        }
    }
}

// ---------------------------------------------------------------------------
// FC via WMMA: Y = prelu(X @ W^T + b). X: MxK f16 (rows 16B-aligned, K%32==0),
// Wt: KxN f16. A loaded as 2x 16B vectors; B gathered with stride N; f32 accum/out.
// ---------------------------------------------------------------------------
template<int NTG>
__global__ __launch_bounds__(256) void fc_wmma_prelu(
    const _Float16* __restrict__ X, const _Float16* __restrict__ Wt,
    const float* __restrict__ bias, const float* __restrict__ alpha,
    float* __restrict__ Y, int M, int N, int K)
{
    int wave = threadIdx.x >> 5;
    int lane = threadIdx.x & 31;
    int hi   = lane >> 4;
    int l15  = lane & 15;
    int ng   = (N >> 4) / NTG;
    int totalTiles = (M >> 4) * ng;
    int tile = blockIdx.x * 8 + wave;
    if (tile >= totalTiles) tile = totalTiles - 1;
    int tm = tile / ng;
    int tg = tile - tm * ng;

    const _Float16* xrow = X + (size_t)(tm * 16 + l15) * K;

    v8f acc[NTG];
#pragma unroll
    for (int t = 0; t < NTG; ++t) acc[t] = (v8f){};

    for (int kc = 0; kc < K; kc += 32) {
        int ka = kc + hi * 8;
        v8h lo = *(const v8h*)(xrow + ka);        // 16B aligned
        v8h hh = *(const v8h*)(xrow + ka + 16);
        v16h a;
#pragma unroll
        for (int i = 0; i < 8; ++i) { a[i] = lo[i]; a[i + 8] = hh[i]; }

        const _Float16* wrow = Wt + (size_t)(kc + hi * 8) * N + tg * (NTG * 16) + l15;
#pragma unroll
        for (int t = 0; t < NTG; ++t) {
            v16h b;
#pragma unroll
            for (int i = 0; i < 8; ++i) {
                b[i]     = wrow[(size_t)i * N + t * 16];
                b[i + 8] = wrow[(size_t)(16 + i) * N + t * 16];
            }
            acc[t] = __builtin_amdgcn_wmma_f32_16x16x32_f16(false, a, false, b,
                                                            (short)0, acc[t], false, false);
        }
    }

#pragma unroll
    for (int t = 0; t < NTG; ++t) {
        int nOut = (tg * NTG + t) * 16 + l15;
        float bn = bias[nOut], an = alpha[nOut];
#pragma unroll
        for (int r = 0; r < 8; ++r) {
            int mrow = tm * 16 + (hi << 3) + r;
            float v = acc[t][r] + bn;
            v = v > 0.f ? v : an * v;
            Y[(size_t)mrow * N + nOut] = v;
        }
    }
}

// ---------------------------------------------------------------------------
// Ceil-mode maxpool on f16; -inf padding realized by bounds-check.
// ---------------------------------------------------------------------------
__global__ __launch_bounds__(256) void maxpool_kernel(
    const _Float16* __restrict__ in, _Float16* __restrict__ out,
    int NC, int Hin, int Win, int Hout, int Wout, int k, int s)
{
    int idx = blockIdx.x * blockDim.x + threadIdx.x;
    int total = NC * Hout * Wout;
    if (idx >= total) return;
    int nc  = idx / (Hout * Wout);
    int rem = idx - nc * (Hout * Wout);
    int ho  = rem / Wout;
    int wo  = rem - ho * Wout;
    float m = -3.4e38f;
    for (int i = 0; i < k; ++i) {
        int h = ho * s + i;
        if (h >= Hin) break;
        for (int j = 0; j < k; ++j) {
            int ww = wo * s + j;
            if (ww >= Win) break;
            m = fmaxf(m, (float)in[((size_t)nc * Hin + h) * Win + ww]);
        }
    }
    out[idx] = (_Float16)m;
}

// out[n][(w*H + h)*C + c] = in[n][c][h][w]   (torch permute(0,3,2,1).flatten)
__global__ __launch_bounds__(256) void repack_whc(
    const _Float16* __restrict__ in, _Float16* __restrict__ out,
    int N, int C, int H, int W)
{
    int idx = blockIdx.x * blockDim.x + threadIdx.x;
    int total = N * C * H * W;
    if (idx >= total) return;
    int n   = idx / (C * H * W);
    int rem = idx - n * (C * H * W);
    int c   = rem / (H * W);
    rem    -= c * (H * W);
    int h   = rem / W;
    int w   = rem - h * W;
    out[(size_t)n * (C * H * W) + ((size_t)w * H + h) * C + c] = in[idx];
}

// ---------------------------------------------------------------------------
// Heads
// ---------------------------------------------------------------------------
__global__ __launch_bounds__(256) void pnet_head_kernel(
    const _Float16* __restrict__ feat,
    const float* __restrict__ w1, const float* __restrict__ b1,   // (2,32)
    const float* __restrict__ w2, const float* __restrict__ b2,   // (4,32)
    float* __restrict__ preg, float* __restrict__ pprob,
    int N, int H, int W)
{
    int idx = blockIdx.x * blockDim.x + threadIdx.x;
    int total = N * H * W;
    if (idx >= total) return;
    int n   = idx / (H * W);
    int rem = idx - n * (H * W);
    int h   = rem / W;
    int ww  = rem - h * W;
    size_t hw = (size_t)H * W;
    size_t base = (size_t)n * 32 * hw + (size_t)h * W + ww;

    float f[32];
#pragma unroll
    for (int c = 0; c < 32; ++c) f[c] = (float)feat[base + (size_t)c * hw];

    float l0 = b1[0], l1 = b1[1];
#pragma unroll
    for (int c = 0; c < 32; ++c) { l0 = fmaf(w1[c], f[c], l0); l1 = fmaf(w1[32 + c], f[c], l1); }
    float mx = fmaxf(l0, l1);
    float e0 = __expf(l0 - mx), e1 = __expf(l1 - mx);
    float inv = 1.f / (e0 + e1);
    size_t ob = (size_t)n * 2 * hw + (size_t)h * W + ww;
    pprob[ob]      = e0 * inv;
    pprob[ob + hw] = e1 * inv;

    size_t rb = (size_t)n * 4 * hw + (size_t)h * W + ww;
#pragma unroll
    for (int o = 0; o < 4; ++o) {
        float v = b2[o];
#pragma unroll
        for (int c = 0; c < 32; ++c) v = fmaf(w2[o * 32 + c], f[c], v);
        preg[rb + (size_t)o * hw] = v;
    }
}

__global__ __launch_bounds__(256) void rnet_head_kernel(
    const float* __restrict__ x,                                   // (N,128)
    const float* __restrict__ w1, const float* __restrict__ b1,    // (2,128)
    const float* __restrict__ w2, const float* __restrict__ b2,    // (4,128)
    float* __restrict__ reg, float* __restrict__ prob, int N, int D)
{
    int n = blockIdx.x * blockDim.x + threadIdx.x;
    if (n >= N) return;
    const float* xr = x + (size_t)n * D;
    float l0 = b1[0], l1 = b1[1];
    for (int k = 0; k < D; ++k) { float xv = xr[k]; l0 = fmaf(w1[k], xv, l0); l1 = fmaf(w1[D + k], xv, l1); }
    float mx = fmaxf(l0, l1);
    float e0 = __expf(l0 - mx), e1 = __expf(l1 - mx);
    float inv = 1.f / (e0 + e1);
    prob[(size_t)n * 2]     = e0 * inv;
    prob[(size_t)n * 2 + 1] = e1 * inv;
    for (int o = 0; o < 4; ++o) {
        float v = b2[o];
        for (int k = 0; k < D; ++k) v = fmaf(w2[o * D + k], xr[k], v);
        reg[(size_t)n * 4 + o] = v;
    }
}

__global__ __launch_bounds__(256) void onet_head_kernel(
    const float* __restrict__ x,                                   // (N,256)
    const float* __restrict__ w1, const float* __restrict__ b1,    // (2,256)
    const float* __restrict__ w2, const float* __restrict__ b2,    // (4,256)
    const float* __restrict__ w3, const float* __restrict__ b3,    // (10,256)
    float* __restrict__ reg, float* __restrict__ lmk, float* __restrict__ prob,
    int N, int D)
{
    int n = blockIdx.x * blockDim.x + threadIdx.x;
    if (n >= N) return;
    const float* xr = x + (size_t)n * D;
    float l0 = b1[0], l1 = b1[1];
    for (int k = 0; k < D; ++k) { float xv = xr[k]; l0 = fmaf(w1[k], xv, l0); l1 = fmaf(w1[D + k], xv, l1); }
    float mx = fmaxf(l0, l1);
    float e0 = __expf(l0 - mx), e1 = __expf(l1 - mx);
    float inv = 1.f / (e0 + e1);
    prob[(size_t)n * 2]     = e0 * inv;
    prob[(size_t)n * 2 + 1] = e1 * inv;
    for (int o = 0; o < 4; ++o) {
        float v = b2[o];
        for (int k = 0; k < D; ++k) v = fmaf(w2[o * D + k], xr[k], v);
        reg[(size_t)n * 4 + o] = v;
    }
    for (int o = 0; o < 10; ++o) {
        float v = b3[o];
        for (int k = 0; k < D; ++k) v = fmaf(w3[o * D + k], xr[k], v);
        lmk[(size_t)n * 10 + o] = v;
    }
}

// ---------------------------------------------------------------------------
static inline int cdiv(long a, long b) { return (int)((a + b - 1) / b); }

extern "C" void kernel_launch(void* const* d_in, const int* in_sizes, int n_in,
                              void* d_out, int out_size, void* d_ws, size_t ws_size,
                              hipStream_t stream) {
    (void)in_sizes; (void)n_in; (void)out_size; (void)ws_size;

    const float* pnet_in = (const float*)d_in[0];   // (8,3,720,1280)
    const float* rnet_in = (const float*)d_in[1];   // (2048,3,24,24)
    const float* onet_in = (const float*)d_in[2];   // (1024,3,48,48)

    const float *pc1w=(const float*)d_in[3],  *pc1b=(const float*)d_in[4],  *pp1=(const float*)d_in[5];
    const float *pc2w=(const float*)d_in[6],  *pc2b=(const float*)d_in[7],  *pp2=(const float*)d_in[8];
    const float *pc3w=(const float*)d_in[9],  *pc3b=(const float*)d_in[10], *pp3=(const float*)d_in[11];
    const float *pc41w=(const float*)d_in[12],*pc41b=(const float*)d_in[13];
    const float *pc42w=(const float*)d_in[14],*pc42b=(const float*)d_in[15];
    const float *rc1w=(const float*)d_in[16], *rc1b=(const float*)d_in[17], *rp1=(const float*)d_in[18];
    const float *rc2w=(const float*)d_in[19], *rc2b=(const float*)d_in[20], *rp2=(const float*)d_in[21];
    const float *rc3w=(const float*)d_in[22], *rc3b=(const float*)d_in[23], *rp3=(const float*)d_in[24];
    const float *rd4w=(const float*)d_in[25], *rd4b=(const float*)d_in[26], *rp4=(const float*)d_in[27];
    const float *rd51w=(const float*)d_in[28],*rd51b=(const float*)d_in[29];
    const float *rd52w=(const float*)d_in[30],*rd52b=(const float*)d_in[31];
    const float *oc1w=(const float*)d_in[32], *oc1b=(const float*)d_in[33], *op1=(const float*)d_in[34];
    const float *oc2w=(const float*)d_in[35], *oc2b=(const float*)d_in[36], *op2=(const float*)d_in[37];
    const float *oc3w=(const float*)d_in[38], *oc3b=(const float*)d_in[39], *op3=(const float*)d_in[40];
    const float *oc4w=(const float*)d_in[41], *oc4b=(const float*)d_in[42], *op4=(const float*)d_in[43];
    const float *od5w=(const float*)d_in[44], *od5b=(const float*)d_in[45], *op5=(const float*)d_in[46];
    const float *od61w=(const float*)d_in[47],*od61b=(const float*)d_in[48];
    const float *od62w=(const float*)d_in[49],*od62b=(const float*)d_in[50];
    const float *od63w=(const float*)d_in[51],*od63b=(const float*)d_in[52];

    float* out = (float*)d_out;
    const size_t P_HW = 355 * 635;
    float* p_reg  = out;
    float* p_prob = p_reg  + (size_t)8 * 4 * P_HW;
    float* r_reg  = p_prob + (size_t)8 * 2 * P_HW;
    float* r_prob = r_reg  + (size_t)2048 * 4;
    float* o_reg  = r_prob + (size_t)2048 * 2;
    float* o_lmk  = o_reg  + (size_t)1024 * 4;
    float* o_prob = o_lmk  + (size_t)1024 * 10;

    // Workspace (all f16 now): A (57,708,800 h) | B (29,108,352 h) | WT
    _Float16* A = (_Float16*)d_ws;
    _Float16* B = A + (size_t)57708800;
    _Float16* WT = B + (size_t)29108352;
    float* Bf = (float*)B;                  // f32 view for FC outputs
    _Float16* wt_pc2 = WT;                  // 96*16
    _Float16* wt_pc3 = WT + 1536;           // 160*32
    _Float16* wt_rc2 = WT + 6656;           // 256*48
    _Float16* wt_rc3 = WT + 18944;          // 192*64
    _Float16* wt_oc2 = WT + 31232;          // 288*64
    _Float16* wt_oc3 = WT + 49664;          // 576*64
    _Float16* wt_oc4 = WT + 86528;          // 256*128
    _Float16* wt_rd4 = WT + 119296;         // 576*128
    _Float16* wt_od5 = WT + 193024;         // 1152*256  (end: 487936 halfs)

    const int T = 256;

    // -------- weight prep (tiny) --------
    prep_weights<<<cdiv(96*16, T),   T, 0, stream>>>(pc2w, wt_pc2, 16, 90, 96);
    prep_weights<<<cdiv(160*32, T),  T, 0, stream>>>(pc3w, wt_pc3, 32, 144, 160);
    prep_weights<<<cdiv(256*48, T),  T, 0, stream>>>(rc2w, wt_rc2, 48, 252, 256);
    prep_weights<<<cdiv(192*64, T),  T, 0, stream>>>(rc3w, wt_rc3, 64, 192, 192);
    prep_weights<<<cdiv(288*64, T),  T, 0, stream>>>(oc2w, wt_oc2, 64, 288, 288);
    prep_weights<<<cdiv(576*64, T),  T, 0, stream>>>(oc3w, wt_oc3, 64, 576, 576);
    prep_weights<<<cdiv(256*128, T), T, 0, stream>>>(oc4w, wt_oc4, 128, 256, 256);
    prep_weights<<<cdiv(576*128, T), T, 0, stream>>>(rd4w, wt_rd4, 128, 576, 576);
    prep_weights<<<cdiv(1152*256,T), T, 0, stream>>>(od5w, wt_od5, 256, 1152, 1152);

    // ================= PNet =================
    {   long total = 8L * 359 * 639;
        conv_std_prelu_pool<3,3,3,10,2,2><<<cdiv(total, T), T, 0, stream>>>(
            pnet_in, pc1w, pc1b, pp1, A, 8, 720, 1280, 718, 1278, 359, 639);
    }
    {   long tiles = cdiv(8L * 357 * 637, 16) * 1;    // NT=1,NTG=1
        conv_wmma_prelu<10,3,3,16,1><<<cdiv(tiles, 8), T, 0, stream>>>(
            A, wt_pc2, pc2b, pp2, B, 8, 359, 639, 357, 637);
    }
    {   long tiles = cdiv(8L * 355 * 635, 16) * 1;    // NT=2,NTG=2
        conv_wmma_prelu<16,3,3,32,2><<<cdiv(tiles, 8), T, 0, stream>>>(
            B, wt_pc3, pc3b, pp3, A, 8, 357, 637, 355, 635);
    }
    {   long total = 8L * 355 * 635;
        pnet_head_kernel<<<cdiv(total, T), T, 0, stream>>>(
            A, pc41w, pc41b, pc42w, pc42b, p_reg, p_prob, 8, 355, 635);
    }

    // ================= RNet =================
    {   long total = 2048L * 11 * 11;
        conv_std_prelu_pool<3,3,3,28,3,2><<<cdiv(total, T), T, 0, stream>>>(
            rnet_in, rc1w, rc1b, rp1, A, 2048, 24, 24, 22, 22, 11, 11);
    }
    {   long tiles = cdiv(2048L * 9 * 9, 16) * 1;     // NT=3,NTG=3
        conv_wmma_prelu<28,3,3,48,3><<<cdiv(tiles, 8), T, 0, stream>>>(
            A, wt_rc2, rc2b, rp2, B, 2048, 11, 11, 9, 9);
    }
    {   long total = 2048L * 48 * 4 * 4;
        maxpool_kernel<<<cdiv(total, T), T, 0, stream>>>(B, A, 2048 * 48, 9, 9, 4, 4, 3, 2);
    }
    {   long tiles = cdiv(2048L * 3 * 3, 16) * 1;     // NT=4,NTG=4
        conv_wmma_prelu<48,2,2,64,4><<<cdiv(tiles, 8), T, 0, stream>>>(
            A, wt_rc3, rc3b, rp3, B, 2048, 4, 4, 3, 3);
    }
    {   long total = 2048L * 64 * 3 * 3;
        repack_whc<<<cdiv(total, T), T, 0, stream>>>(B, A, 2048, 64, 3, 3);
    }
    {   long tiles = (2048L / 16) * 2;                // ng = 8/4 = 2
        fc_wmma_prelu<4><<<cdiv(tiles, 8), T, 0, stream>>>(A, wt_rd4, rd4b, rp4, Bf, 2048, 128, 576);
    }
    rnet_head_kernel<<<cdiv(2048, T), T, 0, stream>>>(Bf, rd51w, rd51b, rd52w, rd52b,
                                                      r_reg, r_prob, 2048, 128);

    // ================= ONet =================
    {   long total = 1024L * 23 * 23;
        conv_std_prelu_pool<3,3,3,32,3,2><<<cdiv(total, T), T, 0, stream>>>(
            onet_in, oc1w, oc1b, op1, A, 1024, 48, 48, 46, 46, 23, 23);
    }
    {   long tiles = cdiv(1024L * 21 * 21, 16) * 1;   // NT=4,NTG=4
        conv_wmma_prelu<32,3,3,64,4><<<cdiv(tiles, 8), T, 0, stream>>>(
            A, wt_oc2, oc2b, op2, B, 1024, 23, 23, 21, 21);
    }
    {   long total = 1024L * 64 * 10 * 10;
        maxpool_kernel<<<cdiv(total, T), T, 0, stream>>>(B, A, 1024 * 64, 21, 21, 10, 10, 3, 2);
    }
    {   long tiles = cdiv(1024L * 8 * 8, 16) * 1;     // NT=4,NTG=4
        conv_wmma_prelu<64,3,3,64,4><<<cdiv(tiles, 8), T, 0, stream>>>(
            A, wt_oc3, oc3b, op3, B, 1024, 10, 10, 8, 8);
    }
    {   long total = 1024L * 64 * 4 * 4;
        maxpool_kernel<<<cdiv(total, T), T, 0, stream>>>(B, A, 1024 * 64, 8, 8, 4, 4, 2, 2);
    }
    {   long tiles = cdiv(1024L * 3 * 3, 16) * 2;     // NT=8,NTG=4 -> NG=2
        conv_wmma_prelu<64,2,2,128,4><<<cdiv(tiles, 8), T, 0, stream>>>(
            A, wt_oc4, oc4b, op4, B, 1024, 4, 4, 3, 3);
    }
    {   long total = 1024L * 128 * 3 * 3;
        repack_whc<<<cdiv(total, T), T, 0, stream>>>(B, A, 1024, 128, 3, 3);
    }
    {   long tiles = (1024L / 16) * 4;                // ng = 16/4 = 4
        fc_wmma_prelu<4><<<cdiv(tiles, 8), T, 0, stream>>>(A, wt_od5, od5b, op5, Bf, 1024, 256, 1152);
    }
    onet_head_kernel<<<cdiv(1024, T), T, 0, stream>>>(Bf, od61w, od61b, od62w, od62b,
                                                      od63w, od63b, o_reg, o_lmk, o_prob,
                                                      1024, 256);
}
